// AdapCNN_Block_28905129902572
// MI455X (gfx1250) — matrix operand order repaired and checked by reading, not compiled, not run on hardware
//
#include <hip/hip_runtime.h>
#include <hip/hip_bf16.h>
#include <stdint.h>

// ---------------------------------------------------------------------------
// AdapCNN block for MI455X (gfx1250, wave32, WMMA + TDM).
//   Dominant op: per-sample 64->64 3x3 conv == 16 GEMMs (M=64,N=126^2,K=576),
//   ~18.7 GFLOP vs ~200 MB HBM traffic -> memory-bound at bf16 WMMA rates.
//   Conv = implicit GEMM on v_wmma_f32_16x16x32_bf16, fp32 accumulate.
//   K is permuted to k' = (kh*3+kw)*64 + c (both operands identically) so
//   every WMMA fragment is two contiguous, 16B-aligned ds_load_b128.
//   Weight tile staging uses the Tensor Data Mover (tensor_load_to_lds,
//   6-arg clang-23 form), overlapped with VALU x-tile staging (fp32->bf16),
//   synchronized with s_wait_tensorcnt + workgroup barrier.
// ---------------------------------------------------------------------------

typedef __attribute__((ext_vector_type(16))) __bf16 v16bf;
typedef __attribute__((ext_vector_type(8)))  float  v8f;
typedef __attribute__((ext_vector_type(4)))  unsigned int v4u;
typedef __attribute__((ext_vector_type(8)))  int  v8i;
typedef __attribute__((ext_vector_type(4)))  int  v4i;

#define IN_NC    64
#define OUT_NC   64
#define KK       576          // 64 * 3 * 3
#define WSTRIDE  584          // padded K' stride (16B rows; 292 dw -> bank spread 36)
#define WPADDED  (OUT_NC * WSTRIDE)   // 37376 bf16 per sample (74752 B)
#define XCELL    72           // bf16 per (row,col) cell: 64 ch + 8 pad (144B, 16B aligned)
#define HW       128
#define OH       126
#define CNN_PARA 36928
#define WPART    36864        // CNN_PARA - OUT_NC

#if __has_builtin(__builtin_amdgcn_tensor_load_to_lds)
#define HAVE_TDM 1
#else
#define HAVE_TDM 0
#endif

static __device__ __forceinline__ unsigned short f2bf(float f) {
    union { float f; unsigned u; } cv; cv.f = f;
    unsigned u = cv.u;
    unsigned r = u + 0x7FFFu + ((u >> 16) & 1u);   // round-to-nearest-even
    return (unsigned short)(r >> 16);
}

union frag16 { uint4 q[2]; v16bf v; };             // 32B = 8 VGPRs

// ---------------------------------------------------------------------------
// Dense layer: out[b][j] = relu(bias[j] + dot(in[b,:], w[j,:])), 512x512.
// ---------------------------------------------------------------------------
__global__ __launch_bounds__(512) void fc_relu_kernel(
    const float* __restrict__ in, const float* __restrict__ w,
    const float* __restrict__ bia, float* __restrict__ outp)
{
    __shared__ float xin[512];
    const int bi = blockIdx.x;
    const int j  = threadIdx.x;
    xin[j] = in[bi * 512 + j];
    __syncthreads();
    float acc = bia[j];
    const float* wr = w + (size_t)j * 512;
    #pragma unroll 8
    for (int k = 0; k < 512; ++k) acc = fmaf(xin[k], wr[k], acc);
    outp[bi * 512 + j] = fmaxf(acc, 0.f);
}

// ---------------------------------------------------------------------------
// Grouped 1x1 layer -> per-sample conv parameters.
// Conv weights emitted bf16, K'-PERMUTED (k' = (kh*3+kw)*64 + c) and padded
// to the conv kernel's LDS row stride, so weight staging is ONE contiguous
// 74752-byte block per sample -> a single TDM descriptor.
// ---------------------------------------------------------------------------
__global__ __launch_bounds__(256) void gen_kernel(
    const float* __restrict__ h2, const float* __restrict__ w3,
    const float* __restrict__ b3, unsigned short* __restrict__ wbf,
    float* __restrict__ biasb)
{
    const int b   = blockIdx.y;
    const int idx = blockIdx.x * 256 + threadIdx.x;
    if (idx >= CNN_PARA) return;
    const int g  = idx / 9232;          // CNN_PARA / 4
    const int oi = idx - g * 9232;
    float acc = b3[idx];
    const float* hp = h2 + b * 512 + g * 128;
    const float* wp = w3 + ((size_t)(g * 9232 + oi)) * 128;
    #pragma unroll 8
    for (int i = 0; i < 128; ++i) acc = fmaf(hp[i], wp[i], acc);
    acc = fmaxf(acc, 0.f);
    if (idx < WPART) {
        int m = idx / KK;
        int k = idx - m * KK;
        int c = k / 9;
        int q = k - c * 9;              // kh*3 + kw
        wbf[(size_t)b * WPADDED + m * WSTRIDE + q * 64 + c] = f2bf(acc);
    } else {
        biasb[b * OUT_NC + (idx - WPART)] = acc;
    }
}

// ---------------------------------------------------------------------------
// Per-sample implicit-GEMM conv via WMMA.
//   grid = (63 row-bands, 16 samples), block = 256 threads = 8 waves.
//   Per WG: M=64, N=256 (2 out rows x 128 cols), K=576 in 18 steps of 32.
//   wave w owns N-tiles {2w,2w+1} and ALL 4 M-tiles (B frags reused 4x).
// ---------------------------------------------------------------------------
__global__ __launch_bounds__(256) void conv_wmma_kernel(
    const float* __restrict__ x, const unsigned short* __restrict__ wbf,
    const float* __restrict__ biasb, float* __restrict__ out)
{
    __shared__ unsigned short wlds[WPADDED];            // 74752 B
    __shared__ unsigned short xlds[4 * 132 * XCELL];    // 76032 B
    __shared__ float bias_lds[OUT_NC];

    const int band = blockIdx.x;        // 0..62 -> output rows 2*band, 2*band+1
    const int b    = blockIdx.y;        // sample
    const int tid  = threadIdx.x;
    const int wave = tid >> 5;
    const int r0   = band * 2;

    const unsigned short* wsrc = wbf + (size_t)b * WPADDED;

#if HAVE_TDM
    // ---- weight tile: one Tensor Data Mover transfer, issued by wave 0 ----
    if (wave == 0) {
        unsigned lds_base = (unsigned)(uintptr_t)(&wlds[0]);
        unsigned long long ga = (unsigned long long)(uintptr_t)wsrc;
        v4u g0;
        g0.x = 1u;                                   // count=1, user descriptor
        g0.y = lds_base;                             // lds_addr (bytes)
        g0.z = (unsigned)ga;                         // global_addr[31:0]
        g0.w = (unsigned)(ga >> 32) | (2u << 30);    // global_addr[56:32] | type=2
        v8i g1;
        g1[0] = 0x10000;                             // data_size=1 (2 bytes)
        g1[1] = (int)((WPADDED & 0xFFFFu) << 16);    // tensor_dim0[15:0]
        g1[2] = (int)((WPADDED >> 16) | (1u << 16)); // tensor_dim0[31:16], tensor_dim1=1
        g1[3] = (int)((unsigned)WPADDED << 16);      // tile_dim0 = 37376
        g1[4] = 1;                                   // tile_dim1=1, tile_dim2=0
        g1[5] = WPADDED;                             // tensor_dim0_stride[31:0]
        g1[6] = 0;
        g1[7] = 0;
        v4i gz4 = {0, 0, 0, 0};                      // D# groups 2/3 (<=2-D tensor)
        v8i gz8 = {0, 0, 0, 0, 0, 0, 0, 0};         // extra words (clang-23 form)
        __builtin_amdgcn_tensor_load_to_lds(g0, g1, gz4, gz4, gz8, 0);
    }
#else
    // ---- fallback: contiguous vectorized copy ----
    {
        const uint2* ws2 = (const uint2*)wsrc;
        uint2* wd2 = (uint2*)wlds;
        for (int i4 = tid; i4 < WPADDED / 4; i4 += 256) wd2[i4] = ws2[i4];
    }
#endif

    if (tid < OUT_NC) bias_lds[tid] = biasb[b * OUT_NC + tid];

    // ---- stage x band (overlaps the TDM): cells [row 0..3][col 0..131] ----
    const float* xsrc = x + (size_t)b * IN_NC * HW * HW;
    for (int i = tid; i < IN_NC * 528; i += 256) {
        int c    = i / 528;             // channel
        int cell = i - c * 528;         // rr*132 + col
        int rr   = cell / 132;
        int col  = cell - rr * 132;
        float v  = 0.f;
        if (col < HW) v = xsrc[((size_t)c * HW + (r0 + rr)) * HW + col];
        xlds[cell * XCELL + c] = f2bf(v);
    }

#if HAVE_TDM
    if (wave == 0) __builtin_amdgcn_s_wait_tensorcnt(0);
#endif
    __syncthreads();

    const int lane  = tid & 31;
    const int lhalf = lane >> 4;
    const int l15   = lane & 15;

    v8f acc[2][4];
    v8f zero = {};
    #pragma unroll
    for (int j = 0; j < 2; ++j)
        #pragma unroll
        for (int mt = 0; mt < 4; ++mt) acc[j][mt] = zero;

    int cellbase[2];
    #pragma unroll
    for (int j = 0; j < 2; ++j) {
        int p = (wave * 2 + j) * 16 + l15;     // N index: r = p>>7, col = p&127
        cellbase[j] = (p >> 7) * 132 + (p & 127);
    }

    for (int ks = 0; ks < 18; ++ks) {
        const int q  = ks >> 1;                // tap index kh*3+kw
        const int kh = q / 3;
        const int kw = q - kh * 3;
        const int c0 = (ks & 1) * 32 + lhalf * 16;

        // ---- B fragments: lane = N col; K = 16 consecutive channels ----
        frag16 bfr[2];
        #pragma unroll
        for (int j = 0; j < 2; ++j) {
            const unsigned short* bp =
                &xlds[(cellbase[j] + kh * 132 + kw) * XCELL + c0];
            bfr[j].q[0] = *reinterpret_cast<const uint4*>(bp);
            bfr[j].q[1] = *reinterpret_cast<const uint4*>(bp + 8);
        }

        // ---- A fragments + WMMA: reuse each B across 4 M-tiles ----
        const int abase = ks * 32 + lhalf * 16;
        #pragma unroll
        for (int mt = 0; mt < 4; ++mt) {
            frag16 af;
            const unsigned short* ap = &wlds[(mt * 16 + l15) * WSTRIDE + abase];
            af.q[0] = *reinterpret_cast<const uint4*>(ap);
            af.q[1] = *reinterpret_cast<const uint4*>(ap + 8);
            #pragma unroll
            for (int j = 0; j < 2; ++j) {
                acc[j][mt] = __builtin_amdgcn_wmma_f32_16x16x32_bf16(
                    false, af.v, false, bfr[j].v, (short)0, acc[j][mt],
                    false, false);
            }
        }
    }

    // ---- store: C/D layout: lane = N col, VGPR i: M = i + (lane>=16?8:0) ----
    #pragma unroll
    for (int j = 0; j < 2; ++j) {
        int p   = (wave * 2 + j) * 16 + l15;
        int r   = p >> 7;
        int col = p & 127;
        if (col < OH) {
            int row = r0 + r;
            #pragma unroll
            for (int mt = 0; mt < 4; ++mt) {
                #pragma unroll
                for (int i = 0; i < 8; ++i) {
                    int m = mt * 16 + lhalf * 8 + i;
                    out[(((size_t)b * OUT_NC + m) * OH + row) * OH + col] =
                        acc[j][mt][i] + bias_lds[m];
                }
            }
        }
    }
}

// ---------------------------------------------------------------------------
extern "C" void kernel_launch(void* const* d_in, const int* in_sizes, int n_in,
                              void* d_out, int out_size, void* d_ws, size_t ws_size,
                              hipStream_t stream) {
    const float* x    = (const float*)d_in[0];
    const float* fcin = (const float*)d_in[1];
    const float* w1   = (const float*)d_in[2];
    const float* b1   = (const float*)d_in[3];
    const float* w2   = (const float*)d_in[4];
    const float* b2   = (const float*)d_in[5];
    const float* w3   = (const float*)d_in[6];
    const float* b3   = (const float*)d_in[7];
    float* out = (float*)d_out;

    char* ws = (char*)d_ws;
    float*          h1    = (float*)(ws);                        // 16*512 f32
    float*          h2    = (float*)(ws + 32768);                // 16*512 f32
    unsigned short* wbf   = (unsigned short*)(ws + 65536);       // 16*37376 bf16, K'-permuted+padded
    float*          biasb = (float*)(ws + 65536 + 16 * WPADDED * 2);  // 16*64 f32

    fc_relu_kernel<<<16, 512, 0, stream>>>(fcin, w1, b1, h1);
    fc_relu_kernel<<<16, 512, 0, stream>>>(h1, w2, b2, h2);
    gen_kernel<<<dim3(145, 16), 256, 0, stream>>>(h2, w3, b3, wbf, biasb);
    conv_wmma_kernel<<<dim3(63, 16), 256, 0, stream>>>(x, wbf, biasb, out);
}
